// CausalSelfAttention_70145405878632
// MI455X (gfx1250) — matrix operand order repaired and checked by reading, not compiled, not run on hardware
//
#include <hip/hip_runtime.h>
#include <hip/hip_bf16.h>

#define BB 2
#define TT 2048
#define CC 2048
#define NHD 16
#define GQ 4
#define NKV 4
#define HD 128

typedef __bf16 bf16;
typedef bf16  v16bf __attribute__((ext_vector_type(16)));
typedef bf16  v8bf  __attribute__((ext_vector_type(8)));
typedef float v8f   __attribute__((ext_vector_type(8)));
typedef unsigned int v4u __attribute__((ext_vector_type(4)));
typedef int   v4i __attribute__((ext_vector_type(4)));
typedef int   v8i __attribute__((ext_vector_type(8)));

#if defined(__gfx1250__) && __has_builtin(__builtin_amdgcn_tensor_load_to_lds) && \
    __has_builtin(__builtin_amdgcn_s_wait_tensorcnt)
#define HAVE_TDM 1
#else
#define HAVE_TDM 0
#endif

// ---------------------------------------------------------------------------
// WMMA helpers (CDNA5 wave32, 16x16x32 bf16 -> f32)
// ---------------------------------------------------------------------------
__device__ __forceinline__ v8f wmma_bf16(v16bf a, v16bf b, v8f c) {
  return __builtin_amdgcn_wmma_f32_16x16x32_bf16(false, a, false, b,
                                                 (short)0, c, false, false);
}

// A fragment (16x32): row M = lane%16; K chunks {0..15} / {16..31}
__device__ __forceinline__ v16bf load_a_frag(const bf16* base, int ld, int lane) {
  int row = lane & 15;
  int ch  = (lane >> 4) << 4;
  const bf16* p = base + (size_t)row * ld + ch;
  union { v16bf v; v8bf h[2]; } u;
  u.h[0] = *(const v8bf*)(p);
  u.h[1] = *(const v8bf*)(p + 8);
  return u.v;
}

// B fragment (32x16): lane holds column N = lane%16 as a contiguous 16-elem
// K chunk (lanes 0-15: K=0..15, lanes 16-31: K=16..31). base = [N][K] rows.
__device__ __forceinline__ v16bf load_b_frag(const bf16* base, int ld, int lane) {
  int col = lane & 15;
  int ch  = (lane >> 4) << 4;
  return *(const v16bf*)(base + (size_t)col * ld + ch);
}

#if HAVE_TDM
// Issue a 2-D TDM tile load: tile_w contiguous elems per row (dim0), tile_h
// rows (dim1) with row stride stride0 (elements, 2-byte data). D# built per
// CDNA5 ISA 8.3/8.4; tensor_dim == tile_dim so no OOB path is ever taken.
// 6-arg builtin: (uint32x4 g0, int32x8 g1, int32x4, int32x4, int32x8, i32 cpol)
__device__ __forceinline__ void tdm_load_2d(unsigned lds_off, const bf16* gptr,
                                            unsigned tile_w, unsigned tile_h,
                                            unsigned stride0) {
  unsigned long long ga = (unsigned long long)(uintptr_t)gptr;
  v4u g0;
  g0[0] = 1u;                                                // count=1, user D#
  g0[1] = lds_off;                                           // lds_addr (bytes)
  g0[2] = (unsigned)(ga & 0xFFFFFFFFu);                      // global_addr lo
  g0[3] = (unsigned)((ga >> 32) & 0x01FFFFFFu) | (2u << 30); // addr hi | type=2
  unsigned long long s0 = (unsigned long long)stride0;
  v8i g1;
  g1[0] = (int)(1u << 16);                                   // wg_mask=0, data_size=1 (2B)
  g1[1] = (int)((tile_w & 0xFFFFu) << 16);                   // tensor_dim0 lo16
  g1[2] = (int)(((tile_w >> 16) & 0xFFFFu) | ((tile_h & 0xFFFFu) << 16));
  g1[3] = (int)(((tile_h >> 16) & 0xFFFFu) | ((tile_w & 0xFFFFu) << 16)); // tile_dim0
  g1[4] = (int)(tile_h & 0xFFFFu);                           // tile_dim1, tile_dim2=0
  g1[5] = (int)(s0 & 0xFFFFFFFFu);                           // dim0_stride lo32
  g1[6] = (int)((s0 >> 32) & 0xFFFFu);                       // dim0_stride hi, dim1_stride=0
  g1[7] = 0;
  v4i z4 = {0, 0, 0, 0};
  v8i z8 = {0, 0, 0, 0, 0, 0, 0, 0};
  __builtin_amdgcn_tensor_load_to_lds(g0, g1, z4, z4, z8, 0);
}
#endif

// ---------------------------------------------------------------------------
// f32 -> bf16 conversion
// ---------------------------------------------------------------------------
__global__ void cvt_f32_bf16(const float* __restrict__ in, bf16* __restrict__ out, int n) {
  int i = blockIdx.x * blockDim.x + threadIdx.x;
  int stride = gridDim.x * blockDim.x;
  for (; i < n; i += stride) out[i] = (bf16)in[i];
}

// ---------------------------------------------------------------------------
// RoPE (reference quirk: angle indexed by HEAD, broadcast over sequence)
// ---------------------------------------------------------------------------
__global__ void rope_kernel(bf16* __restrict__ buf, int npairs, int nh) {
  int idx = blockIdx.x * blockDim.x + threadIdx.x;
  int stride = gridDim.x * blockDim.x;
  for (; idx < npairs; idx += stride) {
    int i = idx & 63;
    int h = (idx >> 6) & (nh - 1);
    float theta = __expf(-(float)i * (0.015625f * 9.210340371976184f)); // 10000^(-i/64)
    float ang = (float)h * theta;
    float s = __sinf(ang), c = __cosf(ang);
    bf16* p = buf + ((size_t)idx << 1);
    float x0 = (float)p[0], x1 = (float)p[1];
    p[0] = (bf16)(x0 * c - x1 * s);
    p[1] = (bf16)(x1 * c + x0 * s);
  }
}

// ---------------------------------------------------------------------------
// GEMM: out = A[M,K] * W[N,K]^T + bias.  MODE 0: bf16 row-major out.
// MODE 1: f32 row-major out.  MODE 2: bf16 out written V-transposed as
// vt[(b*N + n)*T + t]  (m = b*T + t) for the attention P*V loads.
// ---------------------------------------------------------------------------
template <int MODE>
__global__ void __launch_bounds__(256)
gemm_xwT(const bf16* __restrict__ A, const bf16* __restrict__ W,
         const float* __restrict__ bias, void* __restrict__ out,
         int M, int N, int K) {
  const int lane = threadIdx.x & 31;
  const int wave = threadIdx.x >> 5;
  const int m0 = blockIdx.x * 128 + wave * 16;
  const int n0 = blockIdx.y * 64;

  v8f acc0 = 0.f, acc1 = 0.f, acc2 = 0.f, acc3 = 0.f;
  const bf16* arow = A + (size_t)m0 * K;

#pragma unroll 2
  for (int k = 0; k < K; k += 32) {
    v16bf af = load_a_frag(arow + k, K, lane);
    v16bf b0 = load_b_frag(W + (size_t)(n0 +  0) * K + k, K, lane);
    v16bf b1 = load_b_frag(W + (size_t)(n0 + 16) * K + k, K, lane);
    v16bf b2 = load_b_frag(W + (size_t)(n0 + 32) * K + k, K, lane);
    v16bf b3 = load_b_frag(W + (size_t)(n0 + 48) * K + k, K, lane);
    acc0 = wmma_bf16(af, b0, acc0);
    acc1 = wmma_bf16(af, b1, acc1);
    acc2 = wmma_bf16(af, b2, acc2);
    acc3 = wmma_bf16(af, b3, acc3);
  }

  const int ncol = lane & 15;
  const int mr = (lane >> 4) << 3;
  v8f accs[4] = {acc0, acc1, acc2, acc3};
#pragma unroll
  for (int s = 0; s < 4; ++s) {
    float bv = bias[n0 + 16 * s + ncol];
#pragma unroll
    for (int r = 0; r < 8; ++r) {
      float vv = accs[s][r] + bv;
      int m = m0 + mr + r;
      int n = n0 + 16 * s + ncol;
      if (MODE == 1) {
        ((float*)out)[(size_t)m * N + n] = vv;
      } else if (MODE == 0) {
        ((bf16*)out)[(size_t)m * N + n] = (bf16)vv;
      } else {
        int bidx = m >> 11, t = m & (TT - 1);
        ((bf16*)out)[((size_t)(bidx * N + n)) * TT + t] = (bf16)vv;
      }
    }
  }
}

// ---------------------------------------------------------------------------
// Flash attention: block = 64 threads = 2 waves, each owning a 16-row q tile
// of one (b, head). K tile [32x128] and Vt tile [128x32] are staged in LDS
// (TDM async double-buffered when available), shared by both waves.
// ---------------------------------------------------------------------------
__global__ void __launch_bounds__(64)
attn_kernel(const bf16* __restrict__ qb, const bf16* __restrict__ kb,
            const bf16* __restrict__ vtb, bf16* __restrict__ yb) {
  __shared__ __align__(16) bf16 Kt[2][32 * 128];   // [key][hd]
  __shared__ __align__(16) bf16 Vt[2][128 * 32];   // [hd][key]
  __shared__ __align__(16) bf16 Plds[2][16 * 32];

  const int tid = threadIdx.x;
  const int lane = tid & 31;
  const int wv = tid >> 5;
  const int gid = blockIdx.x;
  const int qp = gid & 63;              // T/32 pair-tiles
  const int h  = (gid >> 6) & 15;
  const int b  = gid >> 10;
  const int kv = h >> 2;
  const int q0 = (qp << 5) + (wv << 4);
  const int ldkv = NKV * HD;            // 512

  const bf16* kbase_g  = kb + (size_t)(b * TT) * ldkv + kv * HD;
  const bf16* vtbase_g = vtb + ((size_t)(b * NKV + kv) * HD) * TT;

  // Q fragments (4 K-steps over HD=128)
  const bf16* qbase = qb + (size_t)(b * TT + q0) * CC + h * HD;
  v16bf qf[4];
#pragma unroll
  for (int ks = 0; ks < 4; ++ks) qf[ks] = load_a_frag(qbase + ks * 32, CC, lane);

  v8f acc[8];
#pragma unroll
  for (int t = 0; t < 8; ++t) acc[t] = 0.f;
  float mx[8], ls[8];
#pragma unroll
  for (int r = 0; r < 8; ++r) { mx[r] = -1e30f; ls[r] = 0.f; }

  const float rscale = 0.08838834764831845f;  // 1/sqrt(HD)
  const int ncol = lane & 15;
  const int mr = (lane >> 4) << 3;
  const int ntiles = qp + 1;            // keys 0 .. qp*32+31

  int cur = 0;
#if HAVE_TDM
  if (wv == 0) {
    tdm_load_2d((unsigned)(uintptr_t)&Kt[0][0], kbase_g, 128, 32, ldkv);
    tdm_load_2d((unsigned)(uintptr_t)&Vt[0][0], vtbase_g, 32, 128, TT);
  }
#endif

  for (int kt = 0; kt < ntiles; ++kt) {
    const int k0 = kt << 5;
#if HAVE_TDM
    if (wv == 0) {
      if (kt + 1 < ntiles) {
        int nb = cur ^ 1, nk0 = k0 + 32;
        tdm_load_2d((unsigned)(uintptr_t)&Kt[nb][0],
                    kbase_g + (size_t)nk0 * ldkv, 128, 32, ldkv);
        tdm_load_2d((unsigned)(uintptr_t)&Vt[nb][0],
                    vtbase_g + nk0, 32, 128, TT);
        __builtin_amdgcn_s_wait_tensorcnt(2);
      } else {
        __builtin_amdgcn_s_wait_tensorcnt(0);
      }
    }
    __syncthreads();
#else
    // cooperative fill: 64 threads, 16-byte chunks
    {
      const bf16* src = kbase_g + (size_t)k0 * ldkv;
#pragma unroll 2
      for (int e = tid * 8; e < 32 * 128; e += 64 * 8) {
        int key = e >> 7, hd = e & 127;
        *(v8bf*)&Kt[0][e] = *(const v8bf*)(src + (size_t)key * ldkv + hd);
      }
#pragma unroll 2
      for (int e = tid * 8; e < 128 * 32; e += 64 * 8) {
        int row = e >> 5, c = e & 31;
        *(v8bf*)&Vt[0][e] = *(const v8bf*)(vtbase_g + (size_t)row * TT + k0 + c);
      }
    }
    __syncthreads();
#endif

    // S = Q * K^T (two 16x16 tiles over 32 keys), K fragments from LDS
    v8f s0 = 0.f, s1 = 0.f;
#pragma unroll
    for (int ks = 0; ks < 4; ++ks) {
      v16bf kf0 = load_b_frag(&Kt[cur][ks * 32], 128, lane);
      v16bf kf1 = load_b_frag(&Kt[cur][16 * 128 + ks * 32], 128, lane);
      s0 = wmma_bf16(qf[ks], kf0, s0);
      s1 = wmma_bf16(qf[ks], kf1, s1);
    }

    // online softmax (row M = mr+r lives across 16 lanes at element r)
    float p0[8], p1[8];
#pragma unroll
    for (int r = 0; r < 8; ++r) {
      const int qi = q0 + mr + r;
      float a0 = (k0 + ncol      <= qi) ? s0[r] * rscale : -1e30f;
      float a1 = (k0 + 16 + ncol <= qi) ? s1[r] * rscale : -1e30f;
      float vmax = fmaxf(a0, a1);
      vmax = fmaxf(vmax, __shfl_xor(vmax, 1));
      vmax = fmaxf(vmax, __shfl_xor(vmax, 2));
      vmax = fmaxf(vmax, __shfl_xor(vmax, 4));
      vmax = fmaxf(vmax, __shfl_xor(vmax, 8));
      float mnew = fmaxf(mx[r], vmax);
      float fsc = __expf(mx[r] - mnew);
      float e0 = __expf(a0 - mnew);
      float e1 = __expf(a1 - mnew);
      float rs = e0 + e1;
      rs += __shfl_xor(rs, 1);
      rs += __shfl_xor(rs, 2);
      rs += __shfl_xor(rs, 4);
      rs += __shfl_xor(rs, 8);
      ls[r] = ls[r] * fsc + rs;
      mx[r] = mnew;
      p0[r] = e0;
      p1[r] = e1;
#pragma unroll
      for (int t = 0; t < 8; ++t) acc[t][r] *= fsc;
    }

    // P: C-layout -> LDS -> A-fragment
#pragma unroll
    for (int r = 0; r < 8; ++r) {
      const int row = mr + r;
      Plds[wv][row * 32 + ncol]      = (bf16)p0[r];
      Plds[wv][row * 32 + 16 + ncol] = (bf16)p1[r];
    }
    __syncthreads();
    v16bf pa = load_a_frag(&Plds[wv][0], 32, lane);

    // acc += P * V, V fragments contiguous from transposed LDS tile
#pragma unroll
    for (int t = 0; t < 8; ++t) {
      v16bf vf = load_b_frag(&Vt[cur][(t * 16) * 32], 32, lane);
      acc[t] = wmma_bf16(pa, vf, acc[t]);
    }
    __syncthreads();
#if HAVE_TDM
    cur ^= 1;
#endif
  }

  // normalize, write y[b, q, h, :]
#pragma unroll
  for (int t = 0; t < 8; ++t) {
#pragma unroll
    for (int r = 0; r < 8; ++r) {
      float ov = acc[t][r] / ls[r];
      size_t off = (size_t)(b * TT + q0 + mr + r) * CC + h * HD + t * 16 + ncol;
      yb[off] = (bf16)ov;
    }
  }
}

// ---------------------------------------------------------------------------
// launch
// ---------------------------------------------------------------------------
extern "C" void kernel_launch(void* const* d_in, const int* in_sizes, int n_in,
                              void* d_out, int out_size, void* d_ws, size_t ws_size,
                              hipStream_t stream) {
  (void)in_sizes; (void)n_in; (void)out_size; (void)ws_size;
  const float* x    = (const float*)d_in[0];
  const float* wq_w = (const float*)d_in[1];
  const float* wq_b = (const float*)d_in[2];
  const float* wk_w = (const float*)d_in[3];
  const float* wk_b = (const float*)d_in[4];
  const float* wv_w = (const float*)d_in[5];
  const float* wv_b = (const float*)d_in[6];
  const float* cp_w = (const float*)d_in[7];
  const float* cp_b = (const float*)d_in[8];

  char* ws = (char*)d_ws;
  const size_t MBYTE = 1024 * 1024;
  bf16* xb    = (bf16*)(ws + 0 * MBYTE);   // 16 MB
  bf16* wqb   = (bf16*)(ws + 16 * MBYTE);  //  8 MB
  bf16* wkb   = (bf16*)(ws + 24 * MBYTE);  //  2 MB
  bf16* wvb   = (bf16*)(ws + 26 * MBYTE);  //  2 MB
  bf16* cpb   = (bf16*)(ws + 28 * MBYTE);  //  8 MB
  bf16* qbuf  = (bf16*)(ws + 36 * MBYTE);  // 16 MB
  bf16* kbuf  = (bf16*)(ws + 52 * MBYTE);  //  4 MB
  bf16* vtbuf = (bf16*)(ws + 56 * MBYTE);  //  4 MB  [B][NKV][HD][T]
  bf16* ybuf  = (bf16*)(ws + 60 * MBYTE);  // 16 MB  (76 MB total)

  const int M = BB * TT;  // 4096

  cvt_f32_bf16<<<2048, 256, 0, stream>>>(x,    xb,  M * CC);
  cvt_f32_bf16<<<2048, 256, 0, stream>>>(wq_w, wqb, CC * CC);
  cvt_f32_bf16<<<1024, 256, 0, stream>>>(wk_w, wkb, (CC / GQ) * CC);
  cvt_f32_bf16<<<1024, 256, 0, stream>>>(wv_w, wvb, (CC / GQ) * CC);
  cvt_f32_bf16<<<2048, 256, 0, stream>>>(cp_w, cpb, CC * CC);

  gemm_xwT<0><<<dim3(M / 128, CC / 64), 256, 0, stream>>>(xb, wqb, wq_b, qbuf, M, CC, CC);
  gemm_xwT<0><<<dim3(M / 128, (CC / GQ) / 64), 256, 0, stream>>>(xb, wkb, wk_b, kbuf, M, CC / GQ, CC);
  gemm_xwT<2><<<dim3(M / 128, (CC / GQ) / 64), 256, 0, stream>>>(xb, wvb, wv_b, vtbuf, M, CC / GQ, CC);

  rope_kernel<<<2048, 256, 0, stream>>>(qbuf, M * NHD * (HD / 2), NHD);
  rope_kernel<<<512, 256, 0, stream>>>(kbuf, M * NKV * (HD / 2), NKV);

  attn_kernel<<<BB * NHD * (TT / 32), 64, 0, stream>>>(qbuf, kbuf, vtbuf, ybuf);

  gemm_xwT<1><<<dim3(M / 128, CC / 64), 256, 0, stream>>>(ybuf, cpb, cp_b, d_out, M, CC, CC);
}
